// GlobalClassificationLoss_44513041055965
// MI455X (gfx1250) — matrix operand order repaired and checked by reading, not compiled, not run on hardware
//
#include <hip/hip_runtime.h>
#include <hip/hip_bf16.h>
#include <math.h>

// ---------------------------------------------------------------------------
// Problem constants (from reference): B=8, T=1024, S=2, D=256, N=2000
// M = B*T*S = 16384 rows. GEMM: (16384 x 256) * (256 x 2000), fused logsumexp.
// ---------------------------------------------------------------------------
#define MM      16384
#define DD      256
#define NN      2000
#define NCHUNKS 125          // 2000 / 16, exact — no padding needed

typedef __bf16 v16bf __attribute__((ext_vector_type(16)));
typedef float  v8f   __attribute__((ext_vector_type(8)));

union Frag {                 // one WMMA 16-bit operand fragment: 16 bf16 / lane
    v16bf v;
    uint4 q[2];
};

__device__ __forceinline__ unsigned short f2bf(float f) {
    unsigned int u = __float_as_uint(f);
    unsigned int r = u + 0x7FFFu + ((u >> 16) & 1u);   // round-to-nearest-even
    return (unsigned short)(r >> 16);
}

// WGP(CU)-scope prefetch: pulls the line into ALL cache levels (incl. L0).
// __builtin_prefetch lowers to SE scope (GL2 only); default asm scope is CU.
__device__ __forceinline__ void prefetch_wgp(const void* p) {
    asm volatile("global_prefetch_b8 %0, off" :: "v"(p) : "memory");
}

// ---------------------------------------------------------------------------
// Prep 1: one block per speaker_vector row m.
//   A_bf16[m][d] = bf16(X[m][d]);  x2[m] = ||X[m]||^2 (f32);
//   d_pos[m] = |scale| * ||X[m] - Emb[b][s]||^2 + bias  (exact f32)
// ---------------------------------------------------------------------------
__global__ void prep_rows_kernel(const float* __restrict__ X,
                                 const float* __restrict__ Emb,
                                 const float* __restrict__ scale,
                                 const float* __restrict__ bias,
                                 unsigned short* __restrict__ A,
                                 float* __restrict__ x2,
                                 float* __restrict__ dpos) {
    const int m = blockIdx.x;          // 0..16383
    const int d = threadIdx.x;         // 0..255
    const int s = m & 1;               // S = 2
    const int b = m >> 11;             // m / (T*S) = m / 2048

    const float x = X[(size_t)m * DD + d];
    const float e = Emb[((size_t)b * 2 + s) * DD + d];
    A[(size_t)m * DD + d] = f2bf(x);

    __shared__ float s1[256];
    __shared__ float s2[256];
    const float dx = x - e;
    s1[d] = x * x;
    s2[d] = dx * dx;
    __syncthreads();
#pragma unroll
    for (int o = 128; o > 0; o >>= 1) {
        if (d < o) { s1[d] += s1[d + o]; s2[d] += s2[d + o]; }
        __syncthreads();
    }
    if (d == 0) {
        x2[m]   = s1[0];
        dpos[m] = fabsf(scale[0]) * s2[0] + bias[0];
    }
}

// ---------------------------------------------------------------------------
// Prep 2: one block per embedding row n.
//   E_bf16[n][d] = bf16(E[n][d]);  e2s[n] = |scale| * ||E[n]||^2 (pre-scaled)
// ---------------------------------------------------------------------------
__global__ void prep_emb_kernel(const float* __restrict__ E,
                                const float* __restrict__ scale,
                                unsigned short* __restrict__ Eb,
                                float* __restrict__ e2s) {
    const int n = blockIdx.x;          // 0..1999
    const int d = threadIdx.x;         // 0..255
    const float x = E[(size_t)n * DD + d];
    Eb[(size_t)n * DD + d] = f2bf(x);

    __shared__ float s1[256];
    s1[d] = x * x;
    __syncthreads();
#pragma unroll
    for (int o = 128; o > 0; o >>= 1) {
        if (d < o) s1[d] += s1[d + o];
        __syncthreads();
    }
    if (d == 0) e2s[n] = fabsf(scale[0]) * s1[0];
}

// ---------------------------------------------------------------------------
// Main fused kernel: bf16 WMMA GEMM + online logsumexp over N.
//   grid = 64 blocks, block = 256 threads = 8 wave32.
//   Each wave owns 32 rows (two 16-row tiles), A resident in VGPRs,
//   loops over 125 chunks of 16 columns. K = 256 = 8 x (16x16x32 bf16 WMMA).
// ---------------------------------------------------------------------------
__global__ void __launch_bounds__(256)
loss_main_kernel(const unsigned short* __restrict__ A,   // [16384][256] bf16
                 const unsigned short* __restrict__ Eb,  // [2000][256]  bf16
                 const float* __restrict__ x2,           // [16384]
                 const float* __restrict__ e2s,          // [2000], pre-scaled
                 const float* __restrict__ dpos,         // [16384]
                 const float* __restrict__ scale,
                 const float* __restrict__ bias,
                 float* __restrict__ row_loss) {         // [16384]
    const int wave = threadIdx.x >> 5;
    const int lane = threadIdx.x & 31;
    const int h    = lane >> 4;        // lane half: selects K sub-range
    const int ln   = lane & 15;        // 0..15: A row within tile / B column

    const int row0 = blockIdx.x * 256 + wave * 32;   // this wave's first row

    const float as = fabsf(scale[0]);
    const float bs = bias[0];

    // ---- Load resident A fragments (two 16-row tiles, K = 0..255) ----
    // 16-bit A 16x32 layout: lane holds row M = lane&15;
    //   lanes 0-15:  VGPR0-3 = K k0+0..7,  VGPR4-7 = K k0+16..23
    //   lanes 16-31: VGPR0-3 = K k0+8..15, VGPR4-7 = K k0+24..31
    Frag afrag[2][8];
    float xterm[2][8];
#pragma unroll
    for (int t = 0; t < 2; ++t) {
        const unsigned short* arow = A + (size_t)(row0 + t * 16 + ln) * DD;
#pragma unroll
        for (int k = 0; k < 8; ++k) {
            const int kb = k * 32 + h * 8;
            afrag[t][k].q[0] = *(const uint4*)(arow + kb);
            afrag[t][k].q[1] = *(const uint4*)(arow + kb + 16);
        }
#pragma unroll
        for (int j = 0; j < 8; ++j) {
            const int r = row0 + t * 16 + j + h * 8;   // C-layout row of (j,half)
            xterm[t][j] = -as * x2[r] - bs;
        }
    }

    // ---- Online logsumexp state: one (max,sum) per accumulator element ----
    float mrun[2][8], srun[2][8];
#pragma unroll
    for (int t = 0; t < 2; ++t)
#pragma unroll
        for (int j = 0; j < 8; ++j) { mrun[t][j] = -INFINITY; srun[t][j] = 0.0f; }

    const float twoAs = 2.0f * as;

    // ---- Sweep all N columns in chunks of 16 ----
    for (int chunk = 0; chunk < NCHUNKS; ++chunk) {
        const int n = chunk * 16 + ln;         // this lane's column
        const float e2n = e2s[n];              // already |scale|-scaled

        // B 32x16 layout: lane holds column N = lane&15;
        //   lanes 0-15 hold K k0+0..7,16..23; lanes 16-31 hold K k0+8..15,24..31 (matches A)
        const unsigned short* erow = Eb + (size_t)n * DD;
        Frag bf[8];
#pragma unroll
        for (int k = 0; k < 8; ++k) {
            const int kb = k * 32 + h * 8;
            bf[k].q[0] = *(const uint4*)(erow + kb);
            bf[k].q[1] = *(const uint4*)(erow + kb + 16);
        }

        // WGP-scope prefetch of the next chunk's B rows into L0
        // (each lane's row is 512B -> two cacheline touches).
        if (chunk + 1 < NCHUNKS) {
            const char* enext = (const char*)(erow + 16 * DD);
            prefetch_wgp(enext);
            prefetch_wgp(enext + 256);
        }

#pragma unroll
        for (int t = 0; t < 2; ++t) {
            v8f c = {0.f, 0.f, 0.f, 0.f, 0.f, 0.f, 0.f, 0.f};
#pragma unroll
            for (int k = 0; k < 8; ++k) {
                c = __builtin_amdgcn_wmma_f32_16x16x32_bf16(
                        false, afrag[t][k].v, false, bf[k].v,
                        (short)0, c, false, false);
            }
            // c[j] = xe for (row = row0 + t*16 + j + h*8, col = n)
            // v = -d_all = 2*|s|*xe - |s|*x2 - |s|*e2 - bias
            // Online update with a SINGLE exp and compare/select (no
            // canonicalizing min/max):  s' = s*exp(mo-mn) + exp(v-mn),
            // where one exponent is always 0.
#pragma unroll
            for (int j = 0; j < 8; ++j) {
                const float v   = fmaf(twoAs, c[j], xterm[t][j] - e2n);
                const float mo  = mrun[t][j];
                const bool  nvm = (v > mo);              // new value is the max
                const float mn  = nvm ? v : mo;
                const float u   = (nvm ? mo : v) - mn;   // <= 0
                const float e   = __expf(u);
                const float a   = nvm ? e : 1.0f;
                const float b   = nvm ? 1.0f : e;
                srun[t][j] = fmaf(srun[t][j], a, b);
                mrun[t][j] = mn;
            }
        }
    }

    // ---- Combine the 16 per-column partials across lanes of each half ----
#pragma unroll
    for (int t = 0; t < 2; ++t) {
#pragma unroll
        for (int j = 0; j < 8; ++j) {
            float m = mrun[t][j];
            float s = srun[t][j];
#pragma unroll
            for (int off = 8; off > 0; off >>= 1) {   // stays within 16-lane group
                const float mo = __shfl_xor(m, off, 32);
                const float so = __shfl_xor(s, off, 32);
                const float mn = fmaxf(m, mo);
                s = s * __expf(m - mn) + so * __expf(mo - mn);
                m = mn;
            }
            if (ln == 0) {
                const int r = row0 + t * 16 + j + h * 8;
                row_loss[r] = dpos[r] + m + __logf(s);   // d_pos + logsumexp(-d_all)
            }
        }
    }
}

// ---------------------------------------------------------------------------
// Final mean over all 16384 rows -> scalar (deterministic tree reduce).
// ---------------------------------------------------------------------------
__global__ void reduce_kernel(const float* __restrict__ row_loss,
                              float* __restrict__ out) {
    __shared__ float sm[256];
    float acc = 0.0f;
    for (int i = threadIdx.x; i < MM; i += 256) acc += row_loss[i];
    sm[threadIdx.x] = acc;
    __syncthreads();
#pragma unroll
    for (int o = 128; o > 0; o >>= 1) {
        if (threadIdx.x < o) sm[threadIdx.x] += sm[threadIdx.x + o];
        __syncthreads();
    }
    if (threadIdx.x == 0) out[0] = sm[0] * (1.0f / (float)MM);
}

// ---------------------------------------------------------------------------
// Launch
// ---------------------------------------------------------------------------
extern "C" void kernel_launch(void* const* d_in, const int* in_sizes, int n_in,
                              void* d_out, int out_size, void* d_ws, size_t ws_size,
                              hipStream_t stream) {
    const float* X     = (const float*)d_in[0];   // (8,1024,2,256)
    const float* Emb   = (const float*)d_in[1];   // (8,2,256)
    const float* Eall  = (const float*)d_in[2];   // (2000,256)
    const float* scale = (const float*)d_in[3];   // scalar
    const float* bias  = (const float*)d_in[4];   // scalar
    float* out = (float*)d_out;

    // Workspace layout (all 256B-aligned)
    char* ws = (char*)d_ws;
    unsigned short* A_bf  = (unsigned short*)(ws);                       // 8,388,608 B
    unsigned short* E_bf  = (unsigned short*)(ws + 8388608);             // 1,024,000 B
    float*          x2    = (float*)(ws + 8388608 + 1024000);            //    65,536 B
    float*          e2s   = (float*)(ws + 8388608 + 1024000 + 65536);    //     8,192 B
    float*          dpos  = (float*)(ws + 8388608 + 1024000 + 65536 + 8192);
    float*          rloss = (float*)(ws + 8388608 + 1024000 + 65536 + 8192 + 65536);

    prep_rows_kernel<<<MM, 256, 0, stream>>>(X, Emb, scale, bias, A_bf, x2, dpos);
    prep_emb_kernel<<<NN, 256, 0, stream>>>(Eall, scale, E_bf, e2s);
    loss_main_kernel<<<64, 256, 0, stream>>>(A_bf, E_bf, x2, e2s, dpos, scale, bias, rloss);
    reduce_kernel<<<1, 256, 0, stream>>>(rloss, out);
}
